// DiffusionRNNAgent_18391049961657
// MI455X (gfx1250) — matrix operand order repaired and checked by reading, not compiled, not run on hardware
//
#include <hip/hip_runtime.h>
#include <hip/hip_bf16.h>
#include <cmath>
#include <cstdint>

// ---------------------------------------------------------------------------
// CDNA5 (gfx1250) wave32 WMMA types
// ---------------------------------------------------------------------------
typedef __attribute__((ext_vector_type(16))) __bf16 v16bf;
typedef __attribute__((ext_vector_type(8)))  __bf16 v8bf;
typedef __attribute__((ext_vector_type(8)))  float  v8f;
typedef __attribute__((ext_vector_type(4)))  unsigned int v4u;
typedef __attribute__((ext_vector_type(8)))  int    v8i;
typedef __attribute__((ext_vector_type(4)))  int    v4i;

__device__ __forceinline__ v16bf cat8(v8bf lo, v8bf hi) {
  return __builtin_shufflevector(lo, hi, 0,1,2,3,4,5,6,7,8,9,10,11,12,13,14,15);
}

// A-fragment (16x32 bf16, M x K). lane: m = lane&15, hf = lane>>4.
// Per ISA 7.12.2: lane holds k in [kk+hf*8, +8) and [kk+16+hf*8, +8).
__device__ __forceinline__ v16bf load_a_bf16(const __bf16* A, int ld, int kk,
                                             int m, int hf) {
  const __bf16* r = A + (size_t)m * ld + kk + hf * 8;
  v8bf lo = *(const v8bf*)(r);
  v8bf hi = *(const v8bf*)(r + 16);
  return cat8(lo, hi);
}

// Same A-fragment but converting from fp32 source on the fly.
__device__ __forceinline__ v16bf load_a_f32(const float* A, int ld, int kk,
                                            int m, int hf) {
  const float* r = A + (size_t)m * ld + kk + hf * 8;
  v16bf f;
#pragma unroll
  for (int i = 0; i < 8; ++i) {
    f[i]     = (__bf16)r[i];
    f[i + 8] = (__bf16)r[16 + i];
  }
  return f;
}

// B-fragment (32x16 bf16, K x N) from row-major weight W[N][K].
// lane: n = lane&15 -> row (nbase+n); k in [kk+hf*8, +8) and [kk+16+hf*8, +8).
__device__ __forceinline__ v16bf load_b(const __bf16* W, int ldk, int nbase,
                                        int kk, int n, int hf) {
  const __bf16* r = W + (size_t)(nbase + n) * ldk + kk + hf * 8;
  return cat8(*(const v8bf*)(r), *(const v8bf*)(r + 16));
}

__device__ __forceinline__ v8f wmma_bf16(v16bf a, v16bf b, v8f c) {
  return __builtin_amdgcn_wmma_f32_16x16x32_bf16(false, a, false, b,
                                                 (short)0, c, false, false);
}

__device__ __forceinline__ float mishf(float v) {
  float sp = (v > 20.f) ? v : log1pf(__expf(v));
  return v * tanhf(sp);
}
__device__ __forceinline__ float sigmoidf_(float v) {
  return 1.f / (1.f + __expf(-v));
}

// ---------------------------------------------------------------------------
// Tensor Data Mover: 2D bf16 tile (rows x row_elems) global -> LDS.
// D# per CDNA5 ISA 8.3/8.4: group0 = {count/flags, lds_addr, global_addr,
// addr_hi|type=2}; group1 packs data_size/tensor dims/tile dims/strides.
// ---------------------------------------------------------------------------
#if defined(__has_builtin)
#if __has_builtin(__builtin_amdgcn_tensor_load_to_lds)
#define HAVE_TDM 1
#endif
#endif

#ifdef HAVE_TDM
__device__ __forceinline__ void tdm_load_tile_bf16(const __bf16* gsrc,
                                                   unsigned int lds_off,
                                                   unsigned int rows,
                                                   unsigned int row_elems,
                                                   unsigned int row_stride) {
  unsigned long long ga = (unsigned long long)(uintptr_t)gsrc;
  v4u g0;
  g0[0] = 1u;                                    // count=1 (valid), user mode
  g0[1] = lds_off;                               // lds_addr (bytes)
  g0[2] = (unsigned int)(ga & 0xFFFFFFFFu);      // global_addr[31:0]
  g0[3] = (unsigned int)((ga >> 32) & 0x1FFFFFFu) | (2u << 30);  // [56:32]|type=2
  v8i g1;
  g1[0] = (int)(1u << 16);                                   // wg_mask=0, data_size=1 (2B)
  g1[1] = (int)((row_elems & 0xFFFFu) << 16);                // tensor_dim0[15:0] @63:48
  g1[2] = (int)(((row_elems >> 16) & 0xFFFFu) |              // tensor_dim0[31:16]
                ((rows & 0xFFFFu) << 16));                   // tensor_dim1[15:0]
  g1[3] = (int)(((rows >> 16) & 0xFFFFu) |                   // tensor_dim1[31:16]
                ((row_elems & 0xFFFFu) << 16));              // tile_dim0
  g1[4] = (int)(rows & 0xFFFFu);                             // tile_dim1 (tile_dim2=0)
  g1[5] = (int)row_stride;                                   // tensor_dim0_stride[31:0]
  g1[6] = 0;                                                 // stride hi / dim1_stride lo
  g1[7] = 0;
  v4i z4 = {};
#if __clang_major__ >= 23
  v8i z8 = {};
  __builtin_amdgcn_tensor_load_to_lds(g0, g1, z4, z4, z8, 0);
#else
  __builtin_amdgcn_tensor_load_to_lds(g0, g1, z4, z4, 0);
#endif
}
#endif

// ---------------------------------------------------------------------------
// fp32 -> bf16 weight / activation conversion
// ---------------------------------------------------------------------------
__global__ void k_cvt(const float* __restrict__ src, __bf16* __restrict__ dst,
                      int n) {
  int i = blockIdx.x * blockDim.x + threadIdx.x;
  if (i < n) dst[i] = (__bf16)src[i];
}

// m1_w [256][592] -> padded [256][608]: cols 0..79 copy, 80..95 zero (te pad),
// 96..607 = src cols 80..591 (hidden part).
__global__ void k_cvt_m1(const float* __restrict__ src, __bf16* __restrict__ dst) {
  int i = blockIdx.x * blockDim.x + threadIdx.x;
  if (i >= 256 * 608) return;
  int o = i / 608, k = i % 608;
  float v = 0.f;
  if (k < 80)       v = src[o * 592 + k];
  else if (k >= 96) v = src[o * 592 + (k - 16)];
  dst[i] = (__bf16)v;
}

// ---------------------------------------------------------------------------
// Time-embedding MLP (16 -> 32 mish -> 16), padded to 32 bf16 values.
// ---------------------------------------------------------------------------
__global__ void k_temb(const float* __restrict__ t1w, const float* __restrict__ t1b,
                       const float* __restrict__ t2w, const float* __restrict__ t2b,
                       float tval, __bf16* __restrict__ te32) {
  __shared__ float s1[32];
  int i = threadIdx.x;  // 32 threads
  float emb[16];
  const float lf = -logf(10000.f) / 7.f;
#pragma unroll
  for (int j = 0; j < 8; ++j) {
    float e = tval * __expf(lf * (float)j);
    emb[j]     = __sinf(e);
    emb[j + 8] = __cosf(e);
  }
  float a = t1b[i];
#pragma unroll
  for (int j = 0; j < 16; ++j) a += emb[j] * t1w[i * 16 + j];
  s1[i] = mishf(a);
  __syncthreads();
  if (i < 16) {
    float b = t2b[i];
#pragma unroll
    for (int j = 0; j < 32; ++j) b += s1[j] * t2w[i * 32 + j];
    te32[i] = (__bf16)b;
  } else {
    te32[i] = (__bf16)0.f;
  }
}

// ---------------------------------------------------------------------------
// Kernel 1: x = relu(in @ fc1^T); GRU cell -> h (fp32 to d_out, bf16 to ws)
// Block: 256 threads = 8 waves; 64 batch rows; wave w: M-tile w&3, N-half w>>2
// ---------------------------------------------------------------------------
__global__ void __launch_bounds__(256) k_fc1_gru(
    const float* __restrict__ inputs,   // [B][256]
    const float* __restrict__ hidden,   // [B][512] fp32
    const __bf16* __restrict__ Wfc1,    // [512][256]
    const float* __restrict__ bfc1,
    const __bf16* __restrict__ Wih,     // [1536][512]
    const __bf16* __restrict__ Whh,     // [1536][512]
    const float* __restrict__ bih, const float* __restrict__ bhh,
    const __bf16* __restrict__ hin,     // [B][512] bf16 (converted hidden)
    float* __restrict__ h_out,          // d_out h section [B][512]
    __bf16* __restrict__ hbf) {         // ws new-h bf16 [B][512]
  __shared__ __bf16 s_x[64 * 512];      // relu(fc1) activations, 64 KB

  const int lane = threadIdx.x & 31;
  const int wave = threadIdx.x >> 5;
  const int m = lane & 15, hf = lane >> 4;
  const int mt = wave & 3, nh = wave >> 2;
  const int rowbase = blockIdx.x * 64 + mt * 16;

  // ---- fc1: A fragments register-cached across the wave's 16 n-tiles ----
  const float* Ain = inputs + (size_t)rowbase * 256;
  v16bf afr[8];
#pragma unroll
  for (int k = 0; k < 8; ++k) afr[k] = load_a_f32(Ain, 256, k * 32, m, hf);

  for (int nt = nh * 16; nt < nh * 16 + 16; ++nt) {
    v8f acc = {};
#pragma unroll
    for (int k = 0; k < 8; ++k)
      acc = wmma_bf16(afr[k], load_b(Wfc1, 256, nt * 16, k * 32, m, hf), acc);
    int col = nt * 16 + m;
    float bias = bfc1[col];
#pragma unroll
    for (int r = 0; r < 8; ++r) {
      float v = acc[r] + bias;
      v = v > 0.f ? v : 0.f;
      s_x[(size_t)(mt * 16 + r + 8 * hf) * 512 + col] = (__bf16)v;
    }
  }
  __syncthreads();

  // ---- GRU: gi = x@Wih^T, gh = h@Whh^T, fused gates ----
  const __bf16* Ah = hin + (size_t)rowbase * 512;
  const __bf16* Ax = s_x + (size_t)mt * 16 * 512;

  for (int j = nh * 16; j < nh * 16 + 16; ++j) {
    int cb = j * 16;
    if (j + 1 < nh * 16 + 16) {            // prefetch next gate rows into GL2
      __builtin_prefetch(Wih + (size_t)(cb + 16) * 512, 0, 0);
      __builtin_prefetch(Whh + (size_t)(cb + 16) * 512, 0, 0);
    }
    v8f a_ir = {}, a_iz = {}, a_in = {}, a_hr = {}, a_hz = {}, a_hn = {};
#pragma unroll 4
    for (int k = 0; k < 16; ++k) {
      int kk = k * 32;
      v16bf ax = load_a_bf16(Ax, 512, kk, m, hf);
      v16bf ah = load_a_bf16(Ah, 512, kk, m, hf);
      a_ir = wmma_bf16(ax, load_b(Wih, 512, cb,        kk, m, hf), a_ir);
      a_iz = wmma_bf16(ax, load_b(Wih, 512, 512 + cb,  kk, m, hf), a_iz);
      a_in = wmma_bf16(ax, load_b(Wih, 512, 1024 + cb, kk, m, hf), a_in);
      a_hr = wmma_bf16(ah, load_b(Whh, 512, cb,        kk, m, hf), a_hr);
      a_hz = wmma_bf16(ah, load_b(Whh, 512, 512 + cb,  kk, m, hf), a_hz);
      a_hn = wmma_bf16(ah, load_b(Whh, 512, 1024 + cb, kk, m, hf), a_hn);
    }
    int col = cb + m;
    float bir = bih[col], biz = bih[512 + col], bin_ = bih[1024 + col];
    float bhr = bhh[col], bhz = bhh[512 + col], bhn  = bhh[1024 + col];
#pragma unroll
    for (int r = 0; r < 8; ++r) {
      int row = rowbase + r + 8 * hf;
      float rg = sigmoidf_((a_ir[r] + bir) + (a_hr[r] + bhr));
      float zg = sigmoidf_((a_iz[r] + biz) + (a_hz[r] + bhz));
      float ng = tanhf((a_in[r] + bin_) + rg * (a_hn[r] + bhn));
      float hp = hidden[(size_t)row * 512 + col];
      float hv = (1.f - zg) * ng + zg * hp;
      h_out[(size_t)row * 512 + col] = hv;
      hbf[(size_t)row * 512 + col]  = (__bf16)hv;
    }
  }
}

// ---------------------------------------------------------------------------
// Kernel 2: one reverse-diffusion step (m1/m2/m3/out fused, xt updated in ws).
// h tile staged into LDS once per block via the Tensor Data Mover; the same
// 64 KB of LDS is then reused (union) as the z1 ping/pong buffers.
// ---------------------------------------------------------------------------
__global__ void __launch_bounds__(256) k_step(
    float* __restrict__ xt,             // ws [B][64] fp32, read+write
    const __bf16* __restrict__ hbf,     // ws [B][512] bf16
    const __bf16* __restrict__ te32,    // ws [32] bf16 (padded)
    const __bf16* __restrict__ Wm1, const float* __restrict__ bm1,   // [256][608]
    const __bf16* __restrict__ Wm2, const float* __restrict__ bm2,   // [256][256]
    const __bf16* __restrict__ Wm3, const float* __restrict__ bm3,   // [256][256]
    const __bf16* __restrict__ Wout, const float* __restrict__ bout, // [64][256]
    const float* __restrict__ noise,    // step_noise[s] [B][64]
    float c1, float c2, float sigma) {
  __shared__ union {
    __bf16 h[64 * 512];                          // staged h tile (phase m1)
    struct { __bf16 a[64 * 256]; __bf16 b[64 * 256]; } z;  // z1 ping/pong
  } u;                                           // 64 KB total

  const int lane = threadIdx.x & 31;
  const int wave = threadIdx.x >> 5;
  const int m = lane & 15, hf = lane >> 4;
  const int mt = wave & 3, nh = wave >> 2;
  const int rowbase = blockIdx.x * 64 + mt * 16;

  // ---- stage h tile -> LDS ----
#ifdef HAVE_TDM
  if (wave == 0) {
    tdm_load_tile_bf16(hbf + (size_t)blockIdx.x * 64 * 512,
                       (unsigned int)(uintptr_t)&u.h[0],
                       64u, 512u, 512u);
  }
#else
  {
    const v8bf* src = (const v8bf*)(hbf + (size_t)blockIdx.x * 64 * 512);
    v8bf* dst = (v8bf*)u.h;
    for (int i = threadIdx.x; i < 64 * 512 / 8; i += 256) dst[i] = src[i];
  }
#endif

  // ---- register-cache xt / te A-fragments while the DMA is in flight ----
  const float* Axt = xt + (size_t)rowbase * 64;
  v16bf a_x0 = load_a_f32(Axt, 64, 0,  m, hf);
  v16bf a_x1 = load_a_f32(Axt, 64, 32, m, hf);
  v16bf a_te = load_a_bf16(te32, 0, 0, m, hf);  // row-stride 0: broadcast

#ifdef HAVE_TDM
  if (wave == 0) __builtin_amdgcn_s_wait_tensorcnt(0);
#endif
  __syncthreads();

  // ---- m1: z1 = mish([xt | te | h] @ Wm1^T + b); accs held in registers ----
  const __bf16* Ah = u.h + (size_t)mt * 16 * 512;
  v8f accs[8];
  for (int nt8 = 0; nt8 < 8; ++nt8) {
    int nt = nh * 8 + nt8;
    if (nt8 + 1 < 8)
      __builtin_prefetch(Wm1 + (size_t)(nt + 1) * 16 * 608, 0, 0);
    v8f acc = {};
    acc = wmma_bf16(a_x0, load_b(Wm1, 608, nt * 16, 0,  m, hf), acc);
    acc = wmma_bf16(a_x1, load_b(Wm1, 608, nt * 16, 32, m, hf), acc);
    acc = wmma_bf16(a_te, load_b(Wm1, 608, nt * 16, 64, m, hf), acc);
#pragma unroll 4
    for (int k = 0; k < 16; ++k) {
      v16bf ah = load_a_bf16(Ah, 512, k * 32, m, hf);
      acc = wmma_bf16(ah, load_b(Wm1, 608, nt * 16, 96 + k * 32, m, hf), acc);
    }
    accs[nt8] = acc;
  }
  __syncthreads();                         // all reads of u.h complete

  // write z1 into u.z.a (overlays u.h)
  for (int nt8 = 0; nt8 < 8; ++nt8) {
    int col = (nh * 8 + nt8) * 16 + m;
    float bias = bm1[col];
#pragma unroll
    for (int r = 0; r < 8; ++r)
      u.z.a[(size_t)(mt * 16 + r + 8 * hf) * 256 + col] =
          (__bf16)mishf(accs[nt8][r] + bias);
  }
  __syncthreads();

  // ---- m2: u.z.a -> u.z.b ----
  const __bf16* Aa = u.z.a + (size_t)mt * 16 * 256;
  for (int nt = nh * 8; nt < nh * 8 + 8; ++nt) {
    v8f acc = {};
#pragma unroll
    for (int k = 0; k < 8; ++k)
      acc = wmma_bf16(load_a_bf16(Aa, 256, k * 32, m, hf),
                      load_b(Wm2, 256, nt * 16, k * 32, m, hf), acc);
    int col = nt * 16 + m;
    float bias = bm2[col];
#pragma unroll
    for (int r = 0; r < 8; ++r)
      u.z.b[(size_t)(mt * 16 + r + 8 * hf) * 256 + col] =
          (__bf16)mishf(acc[r] + bias);
  }
  __syncthreads();

  // ---- m3: u.z.b -> u.z.a ----
  const __bf16* Ab = u.z.b + (size_t)mt * 16 * 256;
  for (int nt = nh * 8; nt < nh * 8 + 8; ++nt) {
    v8f acc = {};
#pragma unroll
    for (int k = 0; k < 8; ++k)
      acc = wmma_bf16(load_a_bf16(Ab, 256, k * 32, m, hf),
                      load_b(Wm3, 256, nt * 16, k * 32, m, hf), acc);
    int col = nt * 16 + m;
    float bias = bm3[col];
#pragma unroll
    for (int r = 0; r < 8; ++r)
      u.z.a[(size_t)(mt * 16 + r + 8 * hf) * 256 + col] =
          (__bf16)mishf(acc[r] + bias);
  }
  __syncthreads();

  // ---- out + diffusion update: xt = c1*clip(x0) + c2*xt + sigma*noise ----
  const __bf16* Ao = u.z.a + (size_t)mt * 16 * 256;
  for (int nt = nh * 2; nt < nh * 2 + 2; ++nt) {
    v8f acc = {};
#pragma unroll
    for (int k = 0; k < 8; ++k)
      acc = wmma_bf16(load_a_bf16(Ao, 256, k * 32, m, hf),
                      load_b(Wout, 256, nt * 16, k * 32, m, hf), acc);
    int col = nt * 16 + m;
    float bias = bout[col];
#pragma unroll
    for (int r = 0; r < 8; ++r) {
      int row = rowbase + r + 8 * hf;
      float xr = acc[r] + bias;
      xr = fminf(fmaxf(xr, -10.f), 10.f);
      size_t idx = (size_t)row * 64 + col;
      float xn = c1 * xr + c2 * xt[idx] + sigma * noise[idx];
      xt[idx] = xn;
    }
  }
}

// ---------------------------------------------------------------------------
// Finalize: x_final, q = clip, q_log = log_softmax(q), mask = 0, noise copy
// ---------------------------------------------------------------------------
__global__ void k_final(const float* __restrict__ xt,
                        const float* __restrict__ noiselast,
                        float* __restrict__ x_final, float* __restrict__ q,
                        float* __restrict__ q_log, float* __restrict__ mask,
                        float* __restrict__ noise_out, int B) {
  int row = blockIdx.x * blockDim.x + threadIdx.x;
  if (row >= B) return;
  const float* xr = xt + (size_t)row * 64;
  float mx = -1e30f;
#pragma unroll 4
  for (int j = 0; j < 64; ++j)
    mx = fmaxf(mx, fminf(fmaxf(xr[j], -10.f), 10.f));
  float s = 0.f;
#pragma unroll 4
  for (int j = 0; j < 64; ++j)
    s += __expf(fminf(fmaxf(xr[j], -10.f), 10.f) - mx);
  float ls = mx + logf(s);
#pragma unroll 4
  for (int j = 0; j < 64; ++j) {
    size_t idx = (size_t)row * 64 + j;
    float v = xr[j];
    float qc = fminf(fmaxf(v, -10.f), 10.f);
    x_final[idx]   = v;
    q[idx]         = qc;
    q_log[idx]     = qc - ls;
    noise_out[idx] = noiselast[idx];
  }
  mask[row] = 0.f;
}

// ---------------------------------------------------------------------------
// Host launcher
// ---------------------------------------------------------------------------
extern "C" void kernel_launch(void* const* d_in, const int* in_sizes, int n_in,
                              void* d_out, int out_size, void* d_ws, size_t ws_size,
                              hipStream_t stream) {
  (void)n_in; (void)out_size; (void)ws_size;
  const int B = in_sizes[2] / 64;   // init_noise is [B,64]
  const int T = 10;

  const float* inputs  = (const float*)d_in[0];
  const float* hidden  = (const float*)d_in[1];
  const float* init_nz = (const float*)d_in[2];
  const float* stepnz  = (const float*)d_in[3];
  const float* fc1_w   = (const float*)d_in[4];
  const float* fc1_b   = (const float*)d_in[5];
  const float* w_ih    = (const float*)d_in[6];
  const float* w_hh    = (const float*)d_in[7];
  const float* b_ih    = (const float*)d_in[8];
  const float* b_hh    = (const float*)d_in[9];
  const float* t1_w    = (const float*)d_in[10];
  const float* t1_b    = (const float*)d_in[11];
  const float* t2_w    = (const float*)d_in[12];
  const float* t2_b    = (const float*)d_in[13];
  const float* m1_w    = (const float*)d_in[14];
  const float* m1_b    = (const float*)d_in[15];
  const float* m2_w    = (const float*)d_in[16];
  const float* m2_b    = (const float*)d_in[17];
  const float* m3_w    = (const float*)d_in[18];
  const float* m3_b    = (const float*)d_in[19];
  const float* out_w   = (const float*)d_in[20];
  const float* out_b   = (const float*)d_in[21];

  // ---- workspace layout (256-byte aligned) ----
  char* ws = (char*)d_ws;
  size_t off = 0;
  auto take = [&](size_t bytes) -> size_t {
    size_t o = off; off = (off + bytes + 255) & ~(size_t)255; return o;
  };
  __bf16* Wfc1 = (__bf16*)(ws + take((size_t)512 * 256 * 2));
  __bf16* Wih  = (__bf16*)(ws + take((size_t)1536 * 512 * 2));
  __bf16* Whh  = (__bf16*)(ws + take((size_t)1536 * 512 * 2));
  __bf16* Wm1  = (__bf16*)(ws + take((size_t)256 * 608 * 2));
  __bf16* Wm2  = (__bf16*)(ws + take((size_t)256 * 256 * 2));
  __bf16* Wm3  = (__bf16*)(ws + take((size_t)256 * 256 * 2));
  __bf16* Wout = (__bf16*)(ws + take((size_t)64 * 256 * 2));
  __bf16* te32 = (__bf16*)(ws + take(64));
  __bf16* hin  = (__bf16*)(ws + take((size_t)B * 512 * 2));
  __bf16* hbf  = (__bf16*)(ws + take((size_t)B * 512 * 2));
  float*  xt   = (float*)(ws + take((size_t)B * 64 * 4));

  // ---- output layout (floats, concatenated in return order) ----
  float* o = (float*)d_out;
  float* o_xfinal = o;
  float* o_q      = o + (size_t)B * 64;
  float* o_h      = o + (size_t)B * 64 * 2;
  float* o_qlog   = o_h + (size_t)B * 512;
  float* o_mask   = o_qlog + (size_t)B * 64;
  float* o_noise  = o_mask + (size_t)B;

  // ---- host-side cosine beta schedule (Diffusion-QL) ----
  double ac[11];
  for (int i = 0; i <= 10; ++i) {
    double x = 11.0 * (double)i / 10.0;                  // linspace(0,11,11)
    double c = cos(((x / 11.0) + 0.008) / 1.008 * M_PI * 0.5);
    ac[i] = c * c;
  }
  for (int i = 10; i >= 0; --i) ac[i] /= ac[0];
  double betas[10], alphas[10], acp[10], acp_prev[10];
  double run = 1.0;
  for (int t = 0; t < 10; ++t) {
    double b = 1.0 - ac[t + 1] / ac[t];
    if (b < 0.0) b = 0.0; if (b > 0.999) b = 0.999;
    betas[t] = b; alphas[t] = 1.0 - b;
    acp_prev[t] = run; run *= alphas[t]; acp[t] = run;
  }
  double coef1[10], coef2[10], lv[10];
  for (int t = 0; t < 10; ++t) {
    double pv = betas[t] * (1.0 - acp_prev[t]) / (1.0 - acp[t]);
    lv[t] = log(pv < 1e-20 ? 1e-20 : pv);
    coef1[t] = betas[t] * sqrt(acp_prev[t]) / (1.0 - acp[t]);
    coef2[t] = (1.0 - acp_prev[t]) * sqrt(alphas[t]) / (1.0 - acp[t]);
  }

  // ---- stage 0: convert weights / hidden to bf16 ----
  auto cvt = [&](const float* s, __bf16* d, int n) {
    k_cvt<<<(n + 255) / 256, 256, 0, stream>>>(s, d, n);
  };
  cvt(fc1_w, Wfc1, 512 * 256);
  cvt(w_ih,  Wih,  1536 * 512);
  cvt(w_hh,  Whh,  1536 * 512);
  cvt(m2_w,  Wm2,  256 * 256);
  cvt(m3_w,  Wm3,  256 * 256);
  cvt(out_w, Wout, 64 * 256);
  cvt(hidden, hin, B * 512);
  k_cvt_m1<<<(256 * 608 + 255) / 256, 256, 0, stream>>>(m1_w, Wm1);
  hipMemcpyAsync(xt, init_nz, (size_t)B * 64 * 4, hipMemcpyDeviceToDevice, stream);

  // ---- stage 1: fc1 + GRU ----
  const int nblk = B / 64;
  k_fc1_gru<<<nblk, 256, 0, stream>>>(inputs, hidden, Wfc1, fc1_b, Wih, Whh,
                                      b_ih, b_hh, hin, o_h, hbf);

  // ---- stage 2: 10 reverse diffusion steps (t = 9..0) ----
  for (int s = 0; s < T; ++s) {
    int t = 9 - s;
    float tval  = (float)t;
    float c1    = (float)coef1[t];
    float c2    = (float)coef2[t];
    float sigma = (t > 0) ? (float)exp(0.5 * lv[t]) : 0.0f;
    const float* nz = stepnz + (size_t)s * B * 64;
    k_temb<<<1, 32, 0, stream>>>(t1_w, t1_b, t2_w, t2_b, tval, te32);
    k_step<<<nblk, 256, 0, stream>>>(xt, hbf, te32, Wm1, m1_b, Wm2, m2_b,
                                     Wm3, m3_b, Wout, out_b, nz, c1, c2, sigma);
  }

  // ---- stage 3: finalize outputs ----
  const float* noiselast = stepnz + (size_t)(T - 1) * B * 64;
  k_final<<<(B + 255) / 256, 256, 0, stream>>>(xt, noiselast, o_xfinal, o_q,
                                               o_qlog, o_mask, o_noise, B);
}
